// GATLayer_627065225233
// MI455X (gfx1250) — compile-verified, matrix operand order checked by reading, not run on hardware
//
#include <hip/hip_runtime.h>
#include <hip/hip_bf16.h>

#define B_    8
#define S_    1024
#define T_    1024
#define INF_  128
#define OUTF_ 64
#define H_    4
#define ED_   8

typedef float        v2f   __attribute__((ext_vector_type(2)));
typedef float        v8f   __attribute__((ext_vector_type(8)));
typedef _Float16     v16h  __attribute__((ext_vector_type(16)));
typedef unsigned int u32x4 __attribute__((ext_vector_type(4)));
typedef int          i32x4 __attribute__((ext_vector_type(4)));
typedef int          i32x8 __attribute__((ext_vector_type(8)));

// ---------------------------------------------------------------------------
// Kernel 0: fold edge_W/edge_b into per-head vectors.
//   v[h][d'] = sum_d a_edge[h][d] * edge_W[d][d']   (edge_emb = ef @ edge_W^T + b)
//   c[h]     = sum_d a_edge[h][d] * edge_b[d]
// ---------------------------------------------------------------------------
__global__ void gat_prep(const float* __restrict__ a, const float* __restrict__ eW,
                         const float* __restrict__ eb, float* __restrict__ v,
                         float* __restrict__ c)
{
    const int tid = threadIdx.x;            // 32 threads
    const int h = tid >> 3, dp = tid & 7;
    const float* ae = a + h * (2 * OUTF_ + ED_) + 2 * OUTF_;
    float acc = 0.f;
#pragma unroll
    for (int d = 0; d < ED_; ++d) acc += ae[d] * eW[d * ED_ + dp];
    v[h * ED_ + dp] = acc;
    if (dp == 0) {
        float cc = 0.f;
#pragma unroll
        for (int d = 0; d < ED_; ++d) cc += ae[d] * eb[d];
        c[h] = cc;
    }
}

// ---------------------------------------------------------------------------
// Kernel 1: fp32 projection with V_WMMA_F32_16X16X4_F32.
//   grid (S/16, B, 2): z=0 -> h_source (writes f16 B-fragments + e_s)
//                      z=1 -> h_target (writes e_t only)
// ---------------------------------------------------------------------------
__global__ void gat_project(const float* __restrict__ Xs, const float* __restrict__ Xt,
                            const float* __restrict__ W, const float* __restrict__ a,
                            _Float16* __restrict__ frag, float* __restrict__ e_s,
                            float* __restrict__ e_t)
{
    const int s0   = blockIdx.x * 16;
    const int b    = blockIdx.y;
    const int tgt  = blockIdx.z;
    const float* X = tgt ? Xt : Xs;
    const int lane = threadIdx.x & 31;
    const int h    = threadIdx.x >> 5;
    const int n    = lane & 15;   // A: row M; B/C/D: column N
    const int hi   = lane >> 4;   // half-wave select

    v8f c0 = {}, c1 = {}, c2 = {}, c3 = {};
    const float* Xrow = X + (size_t)(b * S_ + s0 + n) * INF_;
    const float* Wh   = W + (size_t)h * INF_ * OUTF_;

    for (int k = 0; k < INF_; k += 4) {
        const int klo = k + hi * 2;   // f32 A 16x4: lanes0-15 K=0,1 ; lanes16-31 K=2,3
        v2f A; A.x = Xrow[klo]; A.y = Xrow[klo + 1];
        const float* w0 = Wh + (size_t)klo * OUTF_ + n;
        v2f B0, B1, B2, B3;
        B0.x = w0[0];  B0.y = w0[OUTF_];
        B1.x = w0[16]; B1.y = w0[OUTF_ + 16];
        B2.x = w0[32]; B2.y = w0[OUTF_ + 32];
        B3.x = w0[48]; B3.y = w0[OUTF_ + 48];
        c0 = __builtin_amdgcn_wmma_f32_16x16x4_f32(false, A, false, B0, (short)0, c0, false, false);
        c1 = __builtin_amdgcn_wmma_f32_16x16x4_f32(false, A, false, B1, (short)0, c1, false, false);
        c2 = __builtin_amdgcn_wmma_f32_16x16x4_f32(false, A, false, B2, (short)0, c2, false, false);
        c3 = __builtin_amdgcn_wmma_f32_16x16x4_f32(false, A, false, B3, (short)0, c3, false, false);
    }

    // e_{s,t}[b,h,row] = Wh[row,:] . a_vec  (reduce the 16 N-lanes per half-wave)
    const float* avec = a + h * (2 * OUTF_ + ED_) + (tgt ? OUTF_ : 0);
    const float a0 = avec[n], a1 = avec[16 + n], a2 = avec[32 + n], a3 = avec[48 + n];
    float* eout = tgt ? e_t : e_s;
#pragma unroll
    for (int r = 0; r < 8; ++r) {
        float val = c0[r] * a0 + c1[r] * a1 + c2[r] * a2 + c3[r] * a3;
#pragma unroll
        for (int m = 1; m < 16; m <<= 1) val += __shfl_xor(val, m, 16);
        if (n == 0) eout[(size_t)(b * H_ + h) * S_ + s0 + r + hi * 8] = val;
    }

    if (!tgt) {
        // Repack Wh tile (fp32 regs -> f16 LDS -> global) into WMMA-B fragment
        // order for 16x16x32_f16: lane L holds N=L&15, K halves per A-layout.
        __shared__ _Float16 wh[H_][16][72];
#pragma unroll
        for (int r = 0; r < 8; ++r) {
            const int row = r + hi * 8;
            wh[h][row][n]      = (_Float16)c0[r];
            wh[h][row][n + 16] = (_Float16)c1[r];
            wh[h][row][n + 32] = (_Float16)c2[r];
            wh[h][row][n + 48] = (_Float16)c3[r];
        }
        __syncthreads();
        const int sc     = s0 >> 5;         // 32-wide K chunk index
        const int parity = (s0 >> 4) & 1;   // low/high 16 of the K=32 chunk
#pragma unroll
        for (int ot = 0; ot < 4; ++ot) {
            union { _Float16 hn[8]; float4 f4; } u;
#pragma unroll
            for (int j = 0; j < 8; ++j) u.hn[j] = wh[h][hi * 8 + j][ot * 16 + n];
            const size_t base =
                ((((size_t)(b * H_ + h) * (S_ / 32) + sc) * 4 + ot) * 32 + lane) * 16 + parity * 8;
            *(float4*)(frag + base) = u.f4;
        }
    }
}

// ---------------------------------------------------------------------------
// TDM: DMA one 2D ef tile (32 source rows x [16 targets x 8 feats]) into LDS.
// D# group0: count=1 | lds_addr | global_addr(57b) | type=2
// D# group1: data_size=4B, tile_dim0=128, tile_dim1=32, dim0_stride=T_*ED_
// Groups 2/3/4 zero (2D tensor), cpol=0. Tracked with TENSORcnt.
// ---------------------------------------------------------------------------
__device__ __forceinline__ void tdm_load_tile(unsigned int lds_off, const float* gptr)
{
    const unsigned long long ga = (unsigned long long)gptr;
    u32x4 g0;
    g0.x = 1u;                                                  // count=1 (valid user D#)
    g0.y = lds_off;                                             // lds_addr (bytes)
    g0.z = (unsigned int)ga;                                    // global_addr[31:0]
    g0.w = ((unsigned int)(ga >> 32) & 0x01FFFFFFu)             // global_addr[56:32]
         | 0x80000000u;                                         // type=2 ("image")
    i32x8 g1;
    g1[0] = (int)(2u << 16);                                    // data_size=4B, no multicast
    g1[1] = (int)((8192u & 0xFFFFu) << 16);                     // tensor_dim0 = 8192 (lo16)
    g1[2] = (int)(4096u << 16);                                 // dim0 hi=0 | tensor_dim1 lo = 4096
    g1[3] = (int)(128u << 16);                                  // dim1 hi=0 | tile_dim0 = 128
    g1[4] = 32;                                                 // tile_dim1 = 32, tile_dim2 = 0
    g1[5] = (int)(T_ * ED_);                                    // tensor_dim0_stride = 8192 elems
    g1[6] = 0;
    g1[7] = 0;
    const i32x4 gz4 = {0, 0, 0, 0};
    const i32x8 gz8 = {0, 0, 0, 0, 0, 0, 0, 0};
    __builtin_amdgcn_tensor_load_to_lds(g0, g1, gz4, gz4, gz8, 0);
}

// ---------------------------------------------------------------------------
// Kernel 2: fused logits + one-pass softmax + aggregation (flash-style),
// with TDM double-buffered edge_features tiles in LDS.
//   grid (T/16, B), block 128 = 4 waves (one head each).
// ---------------------------------------------------------------------------
__global__ void gat_attend(const float* __restrict__ ef, const unsigned char* __restrict__ adj,
                           const _Float16* __restrict__ frag, const float* __restrict__ e_s,
                           const float* __restrict__ e_t, const float* __restrict__ vv,
                           const float* __restrict__ cc, float* __restrict__ out)
{
    const int t0   = blockIdx.x * 16;
    const int b    = blockIdx.y;
    const int tid  = threadIdx.x;
    const int lane = tid & 31;
    const int h    = tid >> 5;
    const int n    = lane & 15;
    const int hi   = lane >> 4;

    __shared__ __align__(16) float eftile[2][32][16][8];  // 32 KiB TDM double buffer
    __shared__ float msk[32][16];
    __shared__ float esl[H_][32];

    float vh[ED_];
#pragma unroll
    for (int d = 0; d < ED_; ++d) vh[d] = vv[h * ED_ + d];

    const float et = e_t[(size_t)(b * H_ + h) * T_ + t0 + n];
    const float ch = cc[h];

    const unsigned int lb0 = (unsigned int)(unsigned long long)&eftile[0][0][0][0];
    const unsigned int lb1 = (unsigned int)(unsigned long long)&eftile[1][0][0][0];
    const float* gbase = ef + ((size_t)(b * S_) * T_ + t0) * ED_;

    if (h == 0) tdm_load_tile(lb0, gbase);                 // prefetch chunk 0

    v8f acc0 = {}, acc1 = {}, acc2 = {}, acc3 = {};
    float rp = 0.f;   // unnormalized row sum (partial over this lane's K subset)

    for (int scb = 0; scb < S_ / 32; ++scb) {
        const int sbase = scb * 32;

        // ---- stage mask (512 B, one coalesced u32 per thread) + e_s chunk
        {
            const int sl = tid >> 2;
            const int tb = (tid & 3) * 4;
            const unsigned int mv =
                *(const unsigned int*)(adj + (size_t)(b * S_ + sbase + sl) * T_ + t0 + tb);
            msk[sl][tb + 0] = (mv & 0x000000FFu) ? 1.f : 0.f;
            msk[sl][tb + 1] = (mv & 0x0000FF00u) ? 1.f : 0.f;
            msk[sl][tb + 2] = (mv & 0x00FF0000u) ? 1.f : 0.f;
            msk[sl][tb + 3] = (mv & 0xFF000000u) ? 1.f : 0.f;
        }
        esl[h][lane] = e_s[(size_t)(b * H_ + h) * S_ + sbase + lane];

        // ---- wave0: kick off next chunk's DMA, then wait for current chunk
        if (h == 0) {
            if (scb + 1 < S_ / 32) {
                tdm_load_tile(((scb + 1) & 1) ? lb1 : lb0,
                              gbase + (size_t)(scb + 1) * 32 * T_ * ED_);
                __builtin_amdgcn_s_wait_tensorcnt(1);      // current chunk landed
            } else {
                __builtin_amdgcn_s_wait_tensorcnt(0);
            }
        }
        __syncthreads();   // tile + mask + e_s visible to all waves

        // ---- per-head wave phase
        const float* efb = &eftile[scb & 1][0][0][0];
        const _Float16* fb =
            frag + (((size_t)(b * H_ + h) * (S_ / 32) + scb) * 4 * 32 + lane) * 16;
        const v16h B0 = *(const v16h*)(fb);
        const v16h B1 = *(const v16h*)(fb + 512);
        const v16h B2 = *(const v16h*)(fb + 1024);
        const v16h B3 = *(const v16h*)(fb + 1536);

        v16h A;
#pragma unroll
        for (int i = 0; i < 16; ++i) {
            // 16-bit A 16x32 layout: lanes0-15 K=0-7,16-23 ; lanes16-31 K=8-15,24-31
            const int k = ((i < 8) ? i : i + 8) + hi * 8;
            const float4 q0 = *(const float4*)(efb + ((k * 16) + n) * 8);
            const float4 q1 = *(const float4*)(efb + ((k * 16) + n) * 8 + 4);
            const float dd = q0.x * vh[0] + q0.y * vh[1] + q0.z * vh[2] + q0.w * vh[3]
                           + q1.x * vh[4] + q1.y * vh[5] + q1.z * vh[6] + q1.w * vh[7];
            float e = esl[h][k] + et + dd + ch;
            e = (e > 0.f) ? e : 0.2f * e;                 // leaky_relu(0.2)
            const float p = msk[k][n] * __expf(e);        // masked -> exactly 0
            rp += p;
            A[i] = (_Float16)p;
        }
        acc0 = __builtin_amdgcn_wmma_f32_16x16x32_f16(false, A, false, B0, (short)0, acc0, false, false);
        acc1 = __builtin_amdgcn_wmma_f32_16x16x32_f16(false, A, false, B1, (short)0, acc1, false, false);
        acc2 = __builtin_amdgcn_wmma_f32_16x16x32_f16(false, A, false, B2, (short)0, acc2, false, false);
        acc3 = __builtin_amdgcn_wmma_f32_16x16x32_f16(false, A, false, B3, (short)0, acc3, false, false);

        __syncthreads();   // all reads done before next iter's staging/TDM overwrite
    }

    rp += __shfl_xor(rp, 16, 32);        // combine K-split halves -> full row sums
    // lanes 0..15 hold rowsum for t_local = n; C VGPR r is row (r + hi*8)
    float* orow = out + (size_t)b * T_ * (H_ * OUTF_);
#pragma unroll
    for (int r = 0; r < 8; ++r) {
        const float rs  = __shfl(rp, hi * 8 + r, 32);
        const float inv = 1.f / fmaxf(rs, 1e-37f);
        const int t = t0 + r + hi * 8;
        float* op = orow + (size_t)t * (H_ * OUTF_) + h * OUTF_ + n;
        op[0]  = acc0[r] * inv;
        op[16] = acc1[r] * inv;
        op[32] = acc2[r] * inv;
        op[48] = acc3[r] * inv;
    }
}

// ---------------------------------------------------------------------------
extern "C" void kernel_launch(void* const* d_in, const int* in_sizes, int n_in,
                              void* d_out, int out_size, void* d_ws, size_t ws_size,
                              hipStream_t stream)
{
    (void)in_sizes; (void)n_in; (void)out_size; (void)ws_size;
    const float*         h_src = (const float*)d_in[0];
    const float*         h_tgt = (const float*)d_in[1];
    const float*         ef    = (const float*)d_in[2];
    const unsigned char* adj   = (const unsigned char*)d_in[3];
    const float*         W     = (const float*)d_in[4];
    const float*         a     = (const float*)d_in[5];
    const float*         eW    = (const float*)d_in[6];
    const float*         eb    = (const float*)d_in[7];
    float* out = (float*)d_out;

    char* ws = (char*)d_ws;
    _Float16* frag = (_Float16*)ws;                          // 4 MiB f16 B-fragments
    float* e_s = (float*)(ws + 4 * 1024 * 1024);             // 128 KiB
    float* e_t = (float*)(ws + 4 * 1024 * 1024 + 131072);    // 128 KiB
    float* v   = (float*)(ws + 4 * 1024 * 1024 + 262144);    // H*ED floats
    float* c   = v + H_ * ED_;

    gat_prep<<<1, 32, 0, stream>>>(a, eW, eb, v, c);
    gat_project<<<dim3(S_ / 16, B_, 2), 128, 0, stream>>>(h_src, h_tgt, W, a, frag, e_s, e_t);
    gat_attend<<<dim3(T_ / 16, B_), 128, 0, stream>>>(ef, adj, frag, e_s, e_t, v, c, out);
}
